// TemporalGraphNetwork_764504178986
// MI455X (gfx1250) — compile-verified, matrix operand order checked by reading, not run on hardware
//
#include <hip/hip_runtime.h>
#include <hip/hip_bf16.h>
#include <math.h>

// ---- problem constants (from reference) ----
#define S_LEN 16
#define F_INN 64
#define HID   128
#define G3    384          // 3*HID
#define HEADS 4
#define CDIM  32
#define NEG_SLOPE 0.2f

// ---- WMMA types (gfx1250, wave32) ----
typedef __attribute__((ext_vector_type(16))) __bf16 v16bf;
typedef __attribute__((ext_vector_type(8)))  float  v8f;

union BF16Frag { v16bf v; unsigned int u[8]; };

__device__ __forceinline__ unsigned short f2bf(float f) {
  unsigned int u = __float_as_uint(f);
  u += 0x7FFFu + ((u >> 16) & 1u);               // round-to-nearest-even
  return (unsigned short)(u >> 16);
}

// A-fragment (16x32 bf16) from an LDS tile stored row-major [16][stride].
// Lane L<16: row M=L, K = kbase+{0..7,16..23}; lane>=16: K = kbase+{8..15,24..31}.
__device__ __forceinline__ v16bf frag_from_lds(const unsigned short* p, int stride,
                                               int kbase, int lane) {
  int m   = lane & 15;
  int sub = lane >> 4;
  const unsigned short* r = p + m * stride;
  int k0 = kbase + sub * 8;
  int k1 = kbase + 16 + sub * 8;
  BF16Frag f;
#pragma unroll
  for (int i = 0; i < 4; ++i) {
    f.u[i]     = (unsigned int)r[k0 + 2*i] | ((unsigned int)r[k0 + 2*i + 1] << 16);
    f.u[4 + i] = (unsigned int)r[k1 + 2*i] | ((unsigned int)r[k1 + 2*i + 1] << 16);
  }
  return f.v;
}

// B-fragment (32x16 bf16) from global weights W[G][K] row-major (out = x @ W^T,
// so B[k][g] = W[g][k]; a column of B is a contiguous row of W).
__device__ __forceinline__ v16bf frag_from_w(const unsigned short* W, int K,
                                             int gbase, int kbase, int lane) {
  int n   = lane & 15;
  int sub = lane >> 4;
  const unsigned short* r = W + (size_t)(gbase + n) * K;
  uint4 a = *(const uint4*)(r + kbase + sub * 8);        // 8 bf16 = 16B, aligned
  uint4 b = *(const uint4*)(r + kbase + 16 + sub * 8);
  BF16Frag f;
  f.u[0]=a.x; f.u[1]=a.y; f.u[2]=a.z; f.u[3]=a.w;
  f.u[4]=b.x; f.u[5]=b.y; f.u[6]=b.z; f.u[7]=b.w;
  return f.v;
}

__device__ __forceinline__ v8f mma_bf16(v8f c, v16bf a, v16bf b) {
  return __builtin_amdgcn_wmma_f32_16x16x32_bf16(false, a, false, b, (short)0, c,
                                                 false, false);
}

// ---------------- utility kernels ----------------
__global__ void k_fill_f32(float* __restrict__ p, float v, int n) {
  int i = blockIdx.x * blockDim.x + threadIdx.x;
  if (i < n) p[i] = v;
}

__global__ void k_f32_to_bf16(const float* __restrict__ s, unsigned short* __restrict__ d, int n) {
  int i = blockIdx.x * blockDim.x + threadIdx.x;
  if (i < n) d[i] = f2bf(s[i]);
}

// ---------------- GRU step: h = GRUCell(x[:,s,:], h) ----------------
// Fused GEMM: acc_i = x_s @ w_ih^T (K=64), acc_h = h @ w_hh^T (K=128), per
// 16-node row tile, 24 col-tiles of 16 (3H=384). 4 waves x 6 col-tiles each.
__global__ void __launch_bounds__(128)
k_gru_step(const float* __restrict__ xseq, int s, int nNodes,
           const unsigned short* __restrict__ wih, const unsigned short* __restrict__ whh,
           const float* __restrict__ bih, const float* __restrict__ bhh,
           float* __restrict__ h)
{
  __shared__ unsigned short lx[16 * F_INN];   // 2 KB
  __shared__ unsigned short lh[16 * HID];     // 4 KB
  __shared__ float gsum[16 * G3];             // 24 KB : (i+h) sums, all gates
  __shared__ float gin [16 * HID];            // 8 KB  : i-part of n-gate only

  const int tid  = threadIdx.x;
  const int lane = tid & 31;
  const int wv   = tid >> 5;
  const int rb   = blockIdx.x * 16;

  for (int i = tid; i < 16 * F_INN; i += 128) {
    int m = i >> 6, k = i & 63;
    int row = rb + m;
    float v = (row < nNodes) ? xseq[((size_t)row * S_LEN + s) * F_INN + k] : 0.0f;
    lx[i] = f2bf(v);
  }
  for (int i = tid; i < 16 * HID; i += 128) {
    int m = i >> 7, k = i & 127;
    int row = rb + m;
    float v = (row < nNodes) ? h[(size_t)row * HID + k] : 0.0f;
    lh[i] = f2bf(v);
  }
  __syncthreads();

  for (int t = 0; t < 6; ++t) {
    int ct = wv * 6 + t;                       // col tile 0..23
    v8f ai = {};
    v8f ah = {};
#pragma unroll
    for (int kt = 0; kt < 2; ++kt) {           // input part, K = 64
      v16bf a = frag_from_lds(lx, F_INN, kt * 32, lane);
      v16bf b = frag_from_w(wih, F_INN, ct * 16, kt * 32, lane);
      ai = mma_bf16(ai, a, b);
    }
#pragma unroll
    for (int kt = 0; kt < 4; ++kt) {           // hidden part, K = 128
      v16bf a = frag_from_lds(lh, HID, kt * 32, lane);
      v16bf b = frag_from_w(whh, HID, ct * 16, kt * 32, lane);
      ah = mma_bf16(ah, a, b);
    }
    int nn = lane & 15, sub = lane >> 4;
#pragma unroll
    for (int vv = 0; vv < 8; ++vv) {           // C/D layout: M = vv + 8*sub
      int mm = vv + sub * 8;
      gsum[mm * G3 + ct * 16 + nn] = ai[vv] + ah[vv];
      if (ct >= 16) gin[mm * HID + (ct - 16) * 16 + nn] = ai[vv];
    }
  }
  __syncthreads();

  // gate math (f32)
  for (int i = tid; i < 16 * HID; i += 128) {
    int m = i >> 7, c = i & 127;
    int row = rb + m;
    if (row >= nNodes) continue;
    float gr = gsum[m * G3 + c]       + bih[c]       + bhh[c];
    float gz = gsum[m * G3 + 128 + c] + bih[128 + c] + bhh[128 + c];
    float r = 1.0f / (1.0f + __expf(-gr));
    float z = 1.0f / (1.0f + __expf(-gz));
    float i_n = gin[m * HID + c] + bih[256 + c];
    float h_n = (gsum[m * G3 + 256 + c] - gin[m * HID + c]) + bhh[256 + c];
    float nn2 = tanhf(i_n + r * h_n);
    float hold = h[(size_t)row * HID + c];
    h[(size_t)row * HID + c] = (1.0f - z) * nn2 + z * hold;
  }
}

// ---------------- GAT: node projection hproj = x @ W^T ----------------
__global__ void __launch_bounds__(128)
k_gat_proj(const float* __restrict__ x, const unsigned short* __restrict__ w,
           float* __restrict__ hproj, int nNodes)
{
  __shared__ unsigned short lx[16 * HID];
  const int tid = threadIdx.x, lane = tid & 31, wv = tid >> 5;
  const int rb = blockIdx.x * 16;
  for (int i = tid; i < 16 * HID; i += 128) {
    int m = i >> 7, k = i & 127;
    int row = rb + m;
    float v = (row < nNodes) ? x[(size_t)row * HID + k] : 0.0f;
    lx[i] = f2bf(v);
  }
  __syncthreads();
  for (int t = 0; t < 2; ++t) {
    int ct = wv * 2 + t;                       // 8 col tiles, 4 waves x 2
    v8f acc = {};
#pragma unroll
    for (int kt = 0; kt < 4; ++kt) {
      v16bf a = frag_from_lds(lx, HID, kt * 32, lane);
      v16bf b = frag_from_w(w, HID, ct * 16, kt * 32, lane);
      acc = mma_bf16(acc, a, b);
    }
    int nn = lane & 15, sub = lane >> 4;
#pragma unroll
    for (int vv = 0; vv < 8; ++vv) {
      int row = rb + vv + sub * 8;
      if (row < nNodes) hproj[(size_t)row * HID + ct * 16 + nn] = acc[vv];
    }
  }
}

// per-node attention coefficients a_s, a_d
__global__ void k_gat_coef(const float* __restrict__ hproj, const float* __restrict__ asrc,
                           const float* __restrict__ adst, float* __restrict__ as_o,
                           float* __restrict__ ad_o, int nNodes)
{
  int i = blockIdx.x * blockDim.x + threadIdx.x;
  if (i >= nNodes * HEADS) return;
  int n = i >> 2, hd = i & 3;
  const float* hp = hproj + (size_t)n * HID + hd * CDIM;
  float s1 = 0.0f, s2 = 0.0f;
#pragma unroll
  for (int c = 0; c < CDIM; ++c) {
    float v = hp[c];
    s1 += v * asrc[hd * CDIM + c];
    s2 += v * adst[hd * CDIM + c];
  }
  as_o[i] = s1;
  ad_o[i] = s2;
}

__device__ __forceinline__ void atomicMaxF(float* addr, float val) {
  if (val >= 0.0f) atomicMax((int*)addr, __float_as_int(val));
  else             atomicMin((unsigned int*)addr, __float_as_uint(val));
}

// edge pass 1: e = leaky_relu(a_s[src]+a_d[dst]); segment max into m
__global__ void k_edge_logits(const long long* __restrict__ ei, long long nEdges, long long nTot,
                              const float* __restrict__ as_, const float* __restrict__ ad_,
                              float* __restrict__ e, float* __restrict__ m)
{
  long long i = (long long)blockIdx.x * blockDim.x + threadIdx.x;
  if (i >= nTot) return;
  long long src, dst;
  if (i < nEdges) { src = ei[i]; dst = ei[nEdges + i]; }
  else            { src = dst = i - nEdges; }         // self loop
#pragma unroll
  for (int hd = 0; hd < HEADS; ++hd) {
    float v = as_[src * HEADS + hd] + ad_[dst * HEADS + hd];
    v = (v > 0.0f) ? v : NEG_SLOPE * v;
    e[i * HEADS + hd] = v;
    atomicMaxF(&m[dst * HEADS + hd], v);
  }
}

// edge pass 2: ex = exp(e - m[dst]); segment sum into den
__global__ void k_edge_exp(const long long* __restrict__ ei, long long nEdges, long long nTot,
                           const float* __restrict__ m, float* __restrict__ e,
                           float* __restrict__ den)
{
  long long i = (long long)blockIdx.x * blockDim.x + threadIdx.x;
  if (i >= nTot) return;
  long long dst = (i < nEdges) ? ei[nEdges + i] : (i - nEdges);
#pragma unroll
  for (int hd = 0; hd < HEADS; ++hd) {
    float ex = __expf(e[i * HEADS + hd] - m[dst * HEADS + hd]);
    e[i * HEADS + hd] = ex;
    unsafeAtomicAdd(&den[dst * HEADS + hd], ex);
  }
}

// edge pass 3: scatter alpha-weighted messages (32 lanes per edge, coalesced)
__global__ void __launch_bounds__(256)
k_edge_scatter(const long long* __restrict__ ei, long long nEdges, long long nTot,
               const float* __restrict__ e, const float* __restrict__ den,
               const float* __restrict__ hproj, float* __restrict__ xout)
{
  long long eidx = (long long)blockIdx.x * 8 + (threadIdx.x >> 5);
  int j = threadIdx.x & 31;
  if (eidx >= nTot) return;
  long long src, dst;
  if (eidx < nEdges) { src = ei[eidx]; dst = ei[nEdges + eidx]; }
  else               { src = dst = eidx - nEdges; }
#pragma unroll
  for (int hd = 0; hd < HEADS; ++hd) {
    float alpha = e[eidx * HEADS + hd] / (den[dst * HEADS + hd] + 1e-16f);
    float v = hproj[src * HID + hd * CDIM + j] * alpha;
    unsafeAtomicAdd(&xout[dst * HID + hd * CDIM + j], v);
  }
}

__global__ void k_bias_elu(float* __restrict__ x, const float* __restrict__ b, int nNodes) {
  int i = blockIdx.x * blockDim.x + threadIdx.x;
  if (i >= nNodes * HID) return;
  float v = x[i] + b[i & (HID - 1)];
  x[i] = (v > 0.0f) ? v : (__expf(v) - 1.0f);
}

// classifier: out[n] = sigmoid(dot(x[n], w) + b) ; one wave32 per node
__global__ void __launch_bounds__(256)
k_classifier(const float* __restrict__ x, const float* __restrict__ w,
             const float* __restrict__ b, float* __restrict__ out, int nNodes)
{
  int node = blockIdx.x * 8 + (threadIdx.x >> 5);
  int lane = threadIdx.x & 31;
  if (node >= nNodes) return;
  const float4* xp = (const float4*)(x + (size_t)node * HID);
  const float4* wp = (const float4*)w;
  float4 xv = xp[lane];
  float4 wv = wp[lane];
  float s = xv.x * wv.x + xv.y * wv.y + xv.z * wv.z + xv.w * wv.w;
#pragma unroll
  for (int o = 16; o > 0; o >>= 1) s += __shfl_xor(s, o, 32);
  if (lane == 0) out[node] = 1.0f / (1.0f + __expf(-(s + b[0])));
}

// ---------------- launch ----------------
extern "C" void kernel_launch(void* const* d_in, const int* in_sizes, int n_in,
                              void* d_out, int out_size, void* d_ws, size_t ws_size,
                              hipStream_t stream)
{
  const float*     xseq  = (const float*)d_in[0];
  const long long* ei    = (const long long*)d_in[1];   // jnp.int64
  const float*     gw_ih = (const float*)d_in[2];
  const float*     gw_hh = (const float*)d_in[3];
  const float*     gb_ih = (const float*)d_in[4];
  const float*     gb_hh = (const float*)d_in[5];
  const float*     g1w   = (const float*)d_in[6];
  const float*     g1as  = (const float*)d_in[7];
  const float*     g1ad  = (const float*)d_in[8];
  const float*     g1b   = (const float*)d_in[9];
  const float*     g2w   = (const float*)d_in[10];
  const float*     g2as  = (const float*)d_in[11];
  const float*     g2ad  = (const float*)d_in[12];
  const float*     g2b   = (const float*)d_in[13];
  const float*     cw    = (const float*)d_in[14];
  const float*     cb    = (const float*)d_in[15];

  const int nNodes = in_sizes[0] / (S_LEN * F_INN);
  const long long nEdges = (long long)in_sizes[1] / 2;
  const long long nTot = nEdges + nNodes;

  // workspace layout (256B aligned)
  char* ws = (char*)d_ws;
  size_t off = 0;
  auto alloc = [&](size_t bytes) { size_t r = off; off += (bytes + 255) & ~(size_t)255; return r; };
  float* hbuf  = (float*)(ws + alloc((size_t)nNodes * HID * 4));   // GRU h / GAT2 out
  float* xout  = (float*)(ws + alloc((size_t)nNodes * HID * 4));   // GAT1 out / GAT2 in
  float* hproj = (float*)(ws + alloc((size_t)nNodes * HID * 4));
  float* as_b  = (float*)(ws + alloc((size_t)nNodes * HEADS * 4));
  float* ad_b  = (float*)(ws + alloc((size_t)nNodes * HEADS * 4));
  float* m_b   = (float*)(ws + alloc((size_t)nNodes * HEADS * 4));
  float* den_b = (float*)(ws + alloc((size_t)nNodes * HEADS * 4));
  float* e_b   = (float*)(ws + alloc((size_t)nTot * HEADS * 4));
  unsigned short* wih_bf = (unsigned short*)(ws + alloc((size_t)G3 * F_INN * 2));
  unsigned short* whh_bf = (unsigned short*)(ws + alloc((size_t)G3 * HID * 2));
  unsigned short* gw_bf  = (unsigned short*)(ws + alloc((size_t)HID * HID * 2));

  // weights -> bf16 once
  k_f32_to_bf16<<<(G3 * F_INN + 255) / 256, 256, 0, stream>>>(gw_ih, wih_bf, G3 * F_INN);
  k_f32_to_bf16<<<(G3 * HID + 255) / 256, 256, 0, stream>>>(gw_hh, whh_bf, G3 * HID);

  // GRU: h0 = 0, then 16 fused WMMA steps
  k_fill_f32<<<(nNodes * HID + 255) / 256, 256, 0, stream>>>(hbuf, 0.0f, nNodes * HID);
  const int rowBlocks = (nNodes + 15) / 16;
  for (int s = 0; s < S_LEN; ++s)
    k_gru_step<<<rowBlocks, 128, 0, stream>>>(xseq, s, nNodes, wih_bf, whh_bf,
                                              gb_ih, gb_hh, hbuf);

  auto gat_layer = [&](const float* xin, float* xo, const float* w,
                       const float* as_, const float* ad_, const float* b) {
    k_f32_to_bf16<<<(HID * HID + 255) / 256, 256, 0, stream>>>(w, gw_bf, HID * HID);
    k_gat_proj<<<rowBlocks, 128, 0, stream>>>(xin, gw_bf, hproj, nNodes);
    k_gat_coef<<<(nNodes * HEADS + 255) / 256, 256, 0, stream>>>(hproj, as_, ad_,
                                                                 as_b, ad_b, nNodes);
    k_fill_f32<<<(nNodes * HEADS + 255) / 256, 256, 0, stream>>>(m_b, -INFINITY, nNodes * HEADS);
    k_fill_f32<<<(nNodes * HEADS + 255) / 256, 256, 0, stream>>>(den_b, 0.0f, nNodes * HEADS);
    k_fill_f32<<<(nNodes * HID + 255) / 256, 256, 0, stream>>>(xo, 0.0f, nNodes * HID);
    int eb = (int)((nTot + 255) / 256);
    k_edge_logits<<<eb, 256, 0, stream>>>(ei, nEdges, nTot, as_b, ad_b, e_b, m_b);
    k_edge_exp<<<eb, 256, 0, stream>>>(ei, nEdges, nTot, m_b, e_b, den_b);
    int sb = (int)((nTot + 7) / 8);
    k_edge_scatter<<<sb, 256, 0, stream>>>(ei, nEdges, nTot, e_b, den_b, hproj, xo);
    k_bias_elu<<<(nNodes * HID + 255) / 256, 256, 0, stream>>>(xo, b, nNodes);
  };

  gat_layer(hbuf, xout, g1w, g1as, g1ad, g1b);   // GAT1: hbuf -> xout
  gat_layer(xout, hbuf, g2w, g2as, g2ad, g2b);   // GAT2: xout -> hbuf

  k_classifier<<<(nNodes + 7) / 8, 256, 0, stream>>>(hbuf, cw, cb, (float*)d_out, nNodes);
}
